// Encoder_52312701666158
// MI455X (gfx1250) — matrix operand order repaired and checked by reading, not compiled, not run on hardware
//
#include <hip/hip_runtime.h>
#include <hip/hip_bf16.h>
#include <math.h>

typedef __attribute__((ext_vector_type(16))) __bf16 v16bf;
typedef __attribute__((ext_vector_type(8)))  __bf16 v8bf;
typedef __attribute__((ext_vector_type(8)))  float  v8f;

#define NB 16          // batch
#define NN 4096        // nodes
#define DD 64          // feature dim
#define NT 16          // node types
#define NV 1000        // vocab
#define NE 32          // emb dim

// ---------------------------------------------------------------------------
// 1) typeT = type_emb @ W1[0:32,:]  (16x64);  labelT = label_emb @ W1[32:64,:]
// grid = 16+1000 blocks, 64 threads
// ---------------------------------------------------------------------------
__global__ void emb_w_kernel(const float* __restrict__ type_emb,
                             const float* __restrict__ label_emb,
                             const float* __restrict__ W1,
                             float* __restrict__ typeT,
                             float* __restrict__ labelT) {
    int row = blockIdx.x;
    int d = threadIdx.x;            // 0..63
    const float* emb;
    const float* W;
    float* out;
    if (row < NT) {
        emb = type_emb + row * NE;
        W = W1;                     // rows 0..31
        out = typeT + row * DD;
    } else {
        int lr = row - NT;
        emb = label_emb + lr * NE;
        W = W1 + NE * DD;           // rows 32..63
        out = labelT + lr * DD;
    }
    float s = 0.f;
#pragma unroll
    for (int e = 0; e < NE; ++e) s += emb[e] * W[e * DD + d];
    out[d] = s;
}

// ---------------------------------------------------------------------------
// 2) dis[b,n] = rsqrt(max(sum_m adj[b,n,m], 1))   one wave per row
// grid = B*N/8 blocks, 256 threads (8 waves)
// ---------------------------------------------------------------------------
__global__ void deg_kernel(const float* __restrict__ adj,
                           float* __restrict__ dis) {
    int lane = threadIdx.x & 31;
    int wave = threadIdx.x >> 5;
    int row = blockIdx.x * 8 + wave;            // b*N + n
    const float* ar = adj + (size_t)row * NN;
    float s = 0.f;
#pragma unroll 8
    for (int i = 0; i < NN / 32; ++i) s += ar[lane + 32 * i];
#pragma unroll
    for (int off = 16; off > 0; off >>= 1) s += __shfl_xor(s, off, 32);
    if (lane == 0) dis[row] = rsqrtf(fmaxf(s, 1.0f));
}

// ---------------------------------------------------------------------------
// 3) H1[b,n,d] = dis[b,n] * (typeT[types[b,n],d] + labelT[labels[b,n],d])
// (column scale dis[m] folded into H so the agg loop streams raw adj)
// ---------------------------------------------------------------------------
__global__ void gather_kernel(const int* __restrict__ types,
                              const int* __restrict__ labels,
                              const float* __restrict__ typeT,
                              const float* __restrict__ labelT,
                              const float* __restrict__ dis,
                              __bf16* __restrict__ H) {
    int idx = blockIdx.x * blockDim.x + threadIdx.x;   // < B*N*64
    int n = idx >> 6;            // flat b*N+n
    int d = idx & 63;
    float v = typeT[types[n] * DD + d] + labelT[labels[n] * DD + d];
    H[idx] = (__bf16)(v * dis[n]);
}

// ---------------------------------------------------------------------------
// 4) X[b,n,d] = relu( bias[d] + dis[n] * sum_m adj[b,n,m] * H[b,m,d] )
//    where H already carries the dis[m] column scale.
// WMMA bf16 core, software-pipelined with double-buffered LDS.
// block = 256 thr (8 waves) -> 128 rows x 64 cols output tile.
// grid = dim3(N/128, B)
// ---------------------------------------------------------------------------
__global__ void agg_kernel(const float* __restrict__ adj,
                           const float* __restrict__ dis,
                           const __bf16* __restrict__ H,
                           const float* __restrict__ bias,
                           float* __restrict__ X) {
    __shared__ __attribute__((aligned(32))) __bf16 lA[2][128 * 32]; // adj tile bf16, row-major [r][k]
    __shared__ __attribute__((aligned(32))) __bf16 lB[2][64 * 32];  // H tile, column-major [d][m]

    const int b  = blockIdx.y;
    const int n0 = blockIdx.x * 128;
    const int t  = threadIdx.x;
    const int lane = t & 31;
    const int w    = t >> 5;          // wave 0..7
    const int l16  = lane & 15;
    const int half = lane >> 4;       // 0/1
    const int kbase = half * 16;

    const float* disB = dis + b * NN;

    v8f acc[4];
#pragma unroll
    for (int j = 0; j < 4; ++j)
#pragma unroll
        for (int v = 0; v < 8; ++v) acc[j][v] = 0.f;

    // staging thread mapping (A): r = t/2, 16 consecutive k
    const int sr = t >> 1;
    const int skoff = (t & 1) * 16;
    // staging thread mapping (B): m = t%32, 8 consecutive d
    const int sm = t & 31;
    const int sd = (t >> 5) * 8;

    const float*  aBase = adj + ((size_t)(b * NN + n0 + sr)) * NN + skoff;
    const __bf16* hBase = H + ((size_t)(b * NN + sm)) * DD + sd;

    // ---- prologue: prefetch tile 0 into registers ----
    float4 pa[4];
    v8bf   ph;
#pragma unroll
    for (int q = 0; q < 4; ++q) pa[q] = ((const float4*)aBase)[q];
    ph = *(const v8bf*)hBase;

    for (int ko = 0; ko < NN / 32; ++ko) {
        const int buf = ko & 1;

        // ---- commit prefetched tile to LDS ----
        {
            __bf16* dst = &lA[buf][sr * 32 + skoff];
#pragma unroll
            for (int q = 0; q < 4; ++q) {
                dst[q * 4 + 0] = (__bf16)pa[q].x;
                dst[q * 4 + 1] = (__bf16)pa[q].y;
                dst[q * 4 + 2] = (__bf16)pa[q].z;
                dst[q * 4 + 3] = (__bf16)pa[q].w;
            }
#pragma unroll
            for (int j = 0; j < 8; ++j) lB[buf][(sd + j) * 32 + sm] = ph[j];
        }
        __syncthreads();

        // ---- issue next tile's global loads (latency hidden by WMMAs) ----
        if (ko + 1 < NN / 32) {
            const float4* ap = (const float4*)(aBase + (size_t)(ko + 1) * 32);
#pragma unroll
            for (int q = 0; q < 4; ++q) pa[q] = ap[q];
            ph = *(const v8bf*)(hBase + (size_t)(ko + 1) * 32 * DD);
        }

        // ---- A fragment (shared across the wave's 4 d-tiles) ----
        const __bf16* aRow = &lA[buf][(w * 16 + l16) * 32];
        v8bf alo = *(const v8bf*)(aRow + kbase);
        v8bf ahi = *(const v8bf*)(aRow + kbase + 8);
        v16bf afrag;
#pragma unroll
        for (int i = 0; i < 8; ++i) { afrag[i] = alo[i]; afrag[8 + i] = ahi[i]; }

#pragma unroll
        for (int tj = 0; tj < 4; ++tj) {
            v16bf bfrag = *(const v16bf*)&lB[buf][(tj * 16 + l16) * 32 + half * 16];
            acc[tj] = __builtin_amdgcn_wmma_f32_16x16x32_bf16(
                false, afrag, false, bfrag, (short)0, acc[tj], false, false);
        }
    }

    // ---- epilogue: row scale dis[n], bias, relu (D-matrix layout) ----
    float dr[8];
#pragma unroll
    for (int v = 0; v < 8; ++v) dr[v] = disB[n0 + w * 16 + half * 8 + v];
#pragma unroll
    for (int tj = 0; tj < 4; ++tj) {
        int d = tj * 16 + l16;
        float bv = bias[d];
#pragma unroll
        for (int v = 0; v < 8; ++v) {
            int row = n0 + w * 16 + half * 8 + v;
            float val = acc[tj][v] * dr[v] + bv;
            X[((size_t)(b * NN + row)) * DD + d] = val > 0.f ? val : 0.f;
        }
    }
}

// ---------------------------------------------------------------------------
// 5) H2[b,n,d] = dis[b,n] * sum_e X[b,n,e] * W2[e,d]   (bf16 out)
// ---------------------------------------------------------------------------
__global__ void xw_kernel(const float* __restrict__ X,
                          const float* __restrict__ W2,
                          const float* __restrict__ dis,
                          __bf16* __restrict__ H) {
    int idx = blockIdx.x * blockDim.x + threadIdx.x;   // < B*N*64
    int n = idx >> 6;
    int d = idx & 63;
    const float* xr = X + (size_t)n * DD;
    float s = 0.f;
#pragma unroll 8
    for (int e = 0; e < DD; ++e) s += xr[e] * W2[e * DD + d];
    H[idx] = (__bf16)(s * dis[n]);
}

// ---------------------------------------------------------------------------
// 6) pooled = [mean_n X, max_n X]; out = pooled @ Wr^T + br.  one block / batch
// ---------------------------------------------------------------------------
__global__ void pool_kernel(const float* __restrict__ X,
                            const float* __restrict__ Wr,
                            const float* __restrict__ br,
                            float* __restrict__ out) {
    __shared__ float sSum[256];
    __shared__ float sMax[256];
    __shared__ float pooled[2 * DD];
    int b = blockIdx.x;
    int t = threadIdx.x;
    int d = t & 63;
    int c = t >> 6;                  // 4 chunks of 1024 nodes
    float s = 0.f, mx = -1e30f;
    for (int n = c * 1024; n < (c + 1) * 1024; ++n) {
        float v = X[((size_t)(b * NN + n)) * DD + d];
        s += v;
        mx = fmaxf(mx, v);
    }
    sSum[t] = s; sMax[t] = mx;
    __syncthreads();
    if (t < 64) {
        float ss = sSum[t] + sSum[t + 64] + sSum[t + 128] + sSum[t + 192];
        float mm = fmaxf(fmaxf(sMax[t], sMax[t + 64]), fmaxf(sMax[t + 128], sMax[t + 192]));
        pooled[t] = ss * (1.0f / NN);
        pooled[64 + t] = mm;
    }
    __syncthreads();
    if (t < 64) {
        float a = br[t];
        const float* wr = Wr + t * 2 * DD;
#pragma unroll
        for (int j = 0; j < 2 * DD; ++j) a += wr[j] * pooled[j];
        out[b * DD + t] = a;
    }
}

// ---------------------------------------------------------------------------
extern "C" void kernel_launch(void* const* d_in, const int* in_sizes, int n_in,
                              void* d_out, int out_size, void* d_ws, size_t ws_size,
                              hipStream_t stream) {
    const int*   node_types  = (const int*)d_in[0];
    const int*   node_labels = (const int*)d_in[1];
    const float* adj         = (const float*)d_in[2];
    const float* type_emb    = (const float*)d_in[3];
    const float* label_emb   = (const float*)d_in[4];
    const float* W1          = (const float*)d_in[5];
    const float* b1          = (const float*)d_in[6];
    const float* W2          = (const float*)d_in[7];
    const float* b2          = (const float*)d_in[8];
    const float* Wr          = (const float*)d_in[9];
    const float* br          = (const float*)d_in[10];
    float* out = (float*)d_out;

    // workspace layout
    char* ws = (char*)d_ws;
    float*  typeT  = (float*)(ws);                                   // 16*64*4      = 4 KB
    float*  labelT = (float*)(ws + (4096));                          // 1000*64*4    = 256000
    float*  dis    = (float*)(ws + (4096 + 256000));                 // B*N*4        = 262144
    __bf16* Hbuf   = (__bf16*)(ws + (4096 + 256000 + 262144));       // B*N*64*2     = 8388608 (33554432 reserved)
    float*  Xbuf   = (float*)(ws + (4096 + 256000 + 262144 + 33554432)); // B*N*64*4 = 16777216

    // 1) embedding @ W1 tables
    emb_w_kernel<<<NT + NV, DD, 0, stream>>>(type_emb, label_emb, W1, typeT, labelT);
    // 2) degree normalization
    deg_kernel<<<NB * NN / 8, 256, 0, stream>>>(adj, dis);
    // 3) H1 = dis * (x @ W1) via table gather
    gather_kernel<<<NB * NN * DD / 256, 256, 0, stream>>>(node_types, node_labels,
                                                          typeT, labelT, dis, Hbuf);
    // 4) layer 1 aggregation (WMMA)
    agg_kernel<<<dim3(NN / 128, NB), 256, 0, stream>>>(adj, dis, Hbuf, b1, Xbuf);
    // 5) H2 = dis * (X1 @ W2)
    xw_kernel<<<NB * NN * DD / 256, 256, 0, stream>>>(Xbuf, W2, dis, Hbuf);
    // 6) layer 2 aggregation (WMMA)
    agg_kernel<<<dim3(NN / 128, NB), 256, 0, stream>>>(adj, dis, Hbuf, b2, Xbuf);
    // 7) pooling + readout
    pool_kernel<<<NB, 256, 0, stream>>>(Xbuf, Wr, br, out);
}
